// LLFE_87780541596171
// MI455X (gfx1250) — matrix operand order repaired and checked by reading, not compiled, not run on hardware
//
#include <hip/hip_runtime.h>

typedef __attribute__((ext_vector_type(16))) __bf16    v16bf;
typedef __attribute__((ext_vector_type(8)))  float     v8f;
typedef __attribute__((ext_vector_type(4)))  unsigned  v4u;
typedef __attribute__((ext_vector_type(8)))  int       v8i;
typedef __attribute__((ext_vector_type(4)))  int       v4i;

union FragBF { v16bf bf; uint4 u[2]; };

__device__ __forceinline__ unsigned short f2bf(float x) {
  unsigned u = __builtin_bit_cast(unsigned, x);
  u += 0x7FFFu + ((u >> 16) & 1u);          // round-to-nearest-even
  return (unsigned short)(u >> 16);
}

// xor-butterfly across the 16-lane row via DPP ROW_XMASK (pure VALU, no LDS).
template <int MASK>
__device__ __forceinline__ float xshuf(float x) {
  int v = __builtin_amdgcn_update_dpp(0, __builtin_bit_cast(int, x),
                                      0x160 | MASK, 0xf, 0xf, true);
  return __builtin_bit_cast(float, v);
}
__device__ __forceinline__ float hmax16(float x) {
  x = fmaxf(x, xshuf<1>(x));
  x = fmaxf(x, xshuf<2>(x));
  x = fmaxf(x, xshuf<4>(x));
  x = fmaxf(x, xshuf<8>(x));
  return x;
}
__device__ __forceinline__ float hadd16(float x) {
  x += xshuf<1>(x);
  x += xshuf<2>(x);
  x += xshuf<4>(x);
  x += xshuf<8>(x);
  return x;
}

// ---------------------------------------------------------------------------
// TDM: DMA a 64-key x 32-ch bf16 tile (row-major [key][ch]) into LDS with
// 16B padding per 64B row -> 80B LDS pitch (40 ushorts), tracked by TENSORcnt.
// D#: count=1, type=2 (image), data_size=2B, tensor_dim0=32, stride0=32,
// tile_dim0=32, tile_dim1=64, pad_enable, pad_interval=3 (16 DW), amount=3 (4 DW).
// ---------------------------------------------------------------------------
__device__ __forceinline__ void tdm_load_k64(const unsigned short* gptr, unsigned ldsOff) {
  unsigned long long ga = (unsigned long long)(uintptr_t)gptr;
  v4u g0 = { 1u,                                   // count=1, user mode
             ldsOff,                               // lds_addr (bytes)
             (unsigned)ga,                         // global_addr[31:0]
             (unsigned)((ga >> 32) & 0x1FFFFFFull) | 0x80000000u };  // [56:32] | type=2
  v8i g1 = { (int)((1u << 16) | (1u << 20) | (3u << 22) | (3u << 25)),
             (int)(32u << 16),                     // tensor_dim0 = 32
             (int)(4096u << 16),                   // tensor_dim1 = 4096
             (int)(32u << 16),                     // tile_dim0 = 32
             64,                                   // tile_dim1 = 64
             32,                                   // tensor_dim0_stride = 32
             0, 0 };
  v4i z4 = { 0, 0, 0, 0 };
  v8i z8 = { 0, 0, 0, 0, 0, 0, 0, 0 };
  __builtin_amdgcn_tensor_load_to_lds(g0, g1, z4, z4, z8, 0);
}

// ---------------------------------------------------------------------------
// Weight repack: OIHW f32 [64][576] -> bf16 [co][576] (WMMA A-friendly).
// ---------------------------------------------------------------------------
__global__ __launch_bounds__(256)
void LLFE_wcvt(const float* __restrict__ w, unsigned short* __restrict__ wbf, int n) {
  int i = blockIdx.x * 256 + threadIdx.x;
  if (i < n) wbf[i] = f2bf(w[i]);
}

// ---------------------------------------------------------------------------
// Conv2d(3x3,pad=1) + BN + ReLU as implicit GEMM on WMMA (bf16 in, f32 acc).
// ---------------------------------------------------------------------------
#define BM_PITCH 296   // 592 B rows: 16B-aligned, 16 distinct banks across lanes
__global__ __launch_bounds__(256)
void LLFE_conv_gemm(const float* __restrict__ src, const unsigned short* __restrict__ Wbf,
                    const float* __restrict__ g,  const float* __restrict__ be,
                    const float* __restrict__ mn, const float* __restrict__ vr,
                    float* __restrict__ dstF32, unsigned short* __restrict__ dstBf) {
  const int y = blockIdx.x;     // output row 0..63
  const int b = blockIdx.y;     // batch 0..3
  const int tid = threadIdx.x;

  __shared__ __align__(16) unsigned short Bm[64 * BM_PITCH];  // [pix][k-half]
  __shared__ float sh_inv[64], sh_bias[64];

  if (tid < 64) {
    float iv = g[tid] * rsqrtf(vr[tid] + 1e-5f);
    sh_inv[tid]  = iv;
    sh_bias[tid] = be[tid] - mn[tid] * iv;
  }

  const int wave = tid >> 5, lane = tid & 31;
  const int n = lane & 15, hi = lane >> 4;
  const int cobase = (wave >> 1) * 16;
  const int pxbase = (wave & 1) * 32;           // two 16-pix tiles

  v8f c0 = {0.f,0.f,0.f,0.f,0.f,0.f,0.f,0.f};
  v8f c1 = {0.f,0.f,0.f,0.f,0.f,0.f,0.f,0.f};

  const unsigned short* wrow = Wbf + (size_t)(cobase + n) * 576;

  for (int half = 0; half < 2; ++half) {
    const int ci0 = half * 32;
    __syncthreads();   // previous GEMM pass done reading Bm
    // Build im2col half: branchless (clamped loads + cndmask zeroing).
    for (int i = tid; i < 32 * 3 * 64; i += 256) {
      int pix = i & 63;
      int kh  = (i >> 6) % 3;
      int ci  = ci0 + (i >> 6) / 3;
      int sy  = y + kh - 1;
      int syc = sy < 0 ? 0 : (sy > 63 ? 63 : sy);
      const float* srow = src + (((size_t)b * 64 + ci) * 64 + syc) * 64;
      int k = (ci - ci0) * 9 + kh * 3;
      bool yok = (unsigned)sy < 64u;
#pragma unroll
      for (int kw = 0; kw < 3; ++kw) {
        int sx = pix + kw - 1;
        int sxc = sx < 0 ? 0 : (sx > 63 ? 63 : sx);
        float v = srow[sxc];
        v = (yok && (unsigned)sx < 64u) ? v : 0.f;
        Bm[pix * BM_PITCH + k + kw] = f2bf(v);
      }
    }
    __syncthreads();

    // GEMM over this K half: 9 chunks of K=32.
    for (int kc = 0; kc < 288; kc += 32) {
      FragBF af;   // weights: A row m = co, halves = 16 consecutive k
      af.u[0] = *(const uint4*)(wrow + half * 288 + kc + hi * 16);
      af.u[1] = *(const uint4*)(wrow + half * 288 + kc + hi * 16 + 8);
      FragBF b0, b1;  // patches: B col n = pix, halves = 16 consecutive k
      const uint4* p0 = (const uint4*)(&Bm[(pxbase + n) * BM_PITCH + kc + hi * 16]);
      b0.u[0] = p0[0]; b0.u[1] = p0[1];
      const uint4* p1 = (const uint4*)(&Bm[(pxbase + 16 + n) * BM_PITCH + kc + hi * 16]);
      b1.u[0] = p1[0]; b1.u[1] = p1[1];
      c0 = __builtin_amdgcn_wmma_f32_16x16x32_bf16(false, af.bf, false, b0.bf,
                                                   (short)0, c0, false, false);
      c1 = __builtin_amdgcn_wmma_f32_16x16x32_bf16(false, af.bf, false, b1.bf,
                                                   (short)0, c1, false, false);
    }
  }

  // Epilogue: BN + ReLU; write f32 [b][co][q] and bf16 [bh][q][ch].
#pragma unroll
  for (int r = 0; r < 8; ++r) {
    int co = cobase + r + 8 * hi;
    float iv = sh_inv[co], bi = sh_bias[co];
    int bh = b * 2 + (co >> 5), ch = co & 31;
#pragma unroll
    for (int t = 0; t < 2; ++t) {
      float acc = t ? c1[r] : c0[r];
      float yv = fmaxf(acc * iv + bi, 0.f);
      int pix = pxbase + t * 16 + n;
      int q = y * 64 + pix;
      if (dstF32) dstF32[((size_t)(b * 64 + co)) * 4096 + q] = yv;
      dstBf[((size_t)bh * 4096 + q) * 32 + ch] = f2bf(yv);
    }
  }
}

// ---------------------------------------------------------------------------
// Flash attention, HEADS=2, head_dim=32 (== WMMA K), L=4096.
// K/V blocks (64 keys) streamed into LDS by the Tensor Data Mover, double-
// buffered (TENSORcnt). Per block: 4 S-WMMAs + online softmax + 4 O-WMMAs.
// ---------------------------------------------------------------------------
#define KB_PITCH 40   // 80 B rows (TDM-padded): 16B-aligned, conflict-spread
#define VT_PITCH 72   // 144 B rows
#define PT_PITCH 72
__global__ __launch_bounds__(256)
void LLFE_flash_attn(const unsigned short* __restrict__ Qbf,
                     const unsigned short* __restrict__ Kbf,
                     const float* __restrict__ hconv,
                     const float* __restrict__ pscale,
                     float* __restrict__ out) {
  constexpr int L = 4096;
  const int tid  = threadIdx.x;
  const int wave = tid >> 5;
  const int lane = tid & 31;
  const int n    = lane & 15;
  const int hi   = lane >> 4;
  const int bh   = blockIdx.y;
  const int b    = bh >> 1, head = bh & 1;
  const int q0   = blockIdx.x * 128 + wave * 16;
  const float sc = 0.125f * pscale[0];       // c^-0.5 * p, c = 64

  __shared__ __align__(16) unsigned short Kblk[2][64 * KB_PITCH];  // TDM dest
  __shared__ __align__(16) unsigned short Vt[32 * VT_PITCH];       // [ch][key]
  __shared__ __align__(16) unsigned short Pt[8][16 * PT_PITCH];    // per-wave P

  const size_t kvBase = (size_t)bh * L * 32;
  const unsigned short* gK = Kbf + kvBase;

  FragBF qa;
  {
    const uint4* qrow = (const uint4*)(Qbf + kvBase + (size_t)(q0 + n) * 32);
    qa.u[0] = qrow[2 * hi];
    qa.u[1] = qrow[2 * hi + 1];
  }

  v8f o0 = {0.f,0.f,0.f,0.f,0.f,0.f,0.f,0.f};
  v8f o1 = {0.f,0.f,0.f,0.f,0.f,0.f,0.f,0.f};
  float rm[8], ls[8];
#pragma unroll
  for (int r = 0; r < 8; ++r) { rm[r] = -3.4e38f; ls[r] = 0.f; }

  unsigned short* pw = &Pt[wave][0];

  // Prologue: kick off TDM for key block 0 (wave 0 owns the DMA + its counter).
  if (wave == 0)
    tdm_load_k64(gK, (unsigned)(uintptr_t)&Kblk[0][0]);

  for (int kb = 0; kb < L; kb += 64) {
    const int ib = (kb >> 6) & 1;
    if (wave == 0)
      __builtin_amdgcn_s_wait_tensorcnt(0);   // current block landed in LDS
    __syncthreads();                          // publish to all waves
    if (wave == 0 && kb + 64 < L)             // stream next block into other buf
      tdm_load_k64(gK + (size_t)(kb + 64) * 32, (unsigned)(uintptr_t)&Kblk[ib ^ 1][0]);

    const unsigned short* kblk = &Kblk[ib][0];

    // Build transposed V [ch][key] from the staged K block (LDS -> LDS).
    {
      int key = tid >> 2, c8 = (tid & 3) * 8;
      union { uint4 u; unsigned short h[8]; } t;
      t.u = *(const uint4*)(kblk + key * KB_PITCH + c8);
#pragma unroll
      for (int e = 0; e < 8; ++e) Vt[(c8 + e) * VT_PITCH + key] = t.h[e];
    }
    __syncthreads();

    // S = Q K^T : four 16-key tiles, K B-fragments from the staged LDS block.
    v8f z = {0.f,0.f,0.f,0.f,0.f,0.f,0.f,0.f};
    v8f s[4];
#pragma unroll
    for (int t = 0; t < 4; ++t) {
      FragBF kf;
      const uint4* kr = (const uint4*)(kblk + (t * 16 + n) * KB_PITCH + hi * 16);
      kf.u[0] = kr[0];
      kf.u[1] = kr[1];
      s[t] = __builtin_amdgcn_wmma_f32_16x16x32_bf16(false, qa.bf, false, kf.bf,
                                                     (short)0, z, false, false);
    }

    // Online softmax over 64 keys (row m = r + 8*hi across 16 lanes of reg r).
#pragma unroll
    for (int r = 0; r < 8; ++r) {
      float a0 = s[0][r] * sc, a1 = s[1][r] * sc;
      float a2 = s[2][r] * sc, a3 = s[3][r] * sc;
      float mx = fmaxf(fmaxf(a0, a1), fmaxf(a2, a3));
      float mnew = fmaxf(rm[r], hmax16(mx));
      float e0 = __expf(a0 - mnew), e1 = __expf(a1 - mnew);
      float e2 = __expf(a2 - mnew), e3 = __expf(a3 - mnew);
      int m = r + 8 * hi;
      pw[m * PT_PITCH +  0 + n] = f2bf(e0);
      pw[m * PT_PITCH + 16 + n] = f2bf(e1);
      pw[m * PT_PITCH + 32 + n] = f2bf(e2);
      pw[m * PT_PITCH + 48 + n] = f2bf(e3);
      float srow  = hadd16(((e0 + e1) + (e2 + e3)));
      float alpha = __expf(rm[r] - mnew);
      ls[r] = ls[r] * alpha + srow;
      rm[r] = mnew;
      o0[r] *= alpha;
      o1[r] *= alpha;
    }

    // O += P V : P re-read as A-fragments (two K=32 chunks), V from LDS.
#pragma unroll
    for (int c = 0; c < 2; ++c) {
      FragBF pa;
      const uint4* pr = (const uint4*)(pw + n * PT_PITCH + c * 32 + hi * 16);
      pa.u[0] = pr[0];
      pa.u[1] = pr[1];
      FragBF vb0, vb1;
      const uint4* v0 = (const uint4*)(&Vt[n * VT_PITCH + c * 32 + hi * 16]);
      vb0.u[0] = v0[0]; vb0.u[1] = v0[1];
      const uint4* v1 = (const uint4*)(&Vt[(16 + n) * VT_PITCH + c * 32 + hi * 16]);
      vb1.u[0] = v1[0]; vb1.u[1] = v1[1];
      o0 = __builtin_amdgcn_wmma_f32_16x16x32_bf16(false, pa.bf, false, vb0.bf,
                                                   (short)0, o0, false, false);
      o1 = __builtin_amdgcn_wmma_f32_16x16x32_bf16(false, pa.bf, false, vb1.bf,
                                                   (short)0, o1, false, false);
    }
  }

  // Epilogue: normalize, transpose back to [b][c][hw], add v_h.
#pragma unroll
  for (int r = 0; r < 8; ++r) {
    float inv = 1.0f / ls[r];
    int m = r + 8 * hi;
    int q = q0 + m;
    size_t idx0 = ((size_t)(b * 64 + head * 32 + n)) * L + q;
    size_t idx1 = idx0 + (size_t)16 * L;
    out[idx0] = o0[r] * inv + hconv[idx0];
    out[idx1] = o1[r] * inv + hconv[idx1];
  }
}

extern "C" void kernel_launch(void* const* d_in, const int* in_sizes, int n_in,
                              void* d_out, int out_size, void* d_ws, size_t ws_size,
                              hipStream_t stream) {
  const float* low  = (const float*)d_in[0];
  const float* high = (const float*)d_in[1];
  const float* wH = (const float*)d_in[2];
  const float* gH = (const float*)d_in[3];
  const float* bH = (const float*)d_in[4];
  const float* mH = (const float*)d_in[5];
  const float* vH = (const float*)d_in[6];
  const float* wL = (const float*)d_in[7];
  const float* gL = (const float*)d_in[8];
  const float* bL = (const float*)d_in[9];
  const float* mL = (const float*)d_in[10];
  const float* vL = (const float*)d_in[11];
  const float* p  = (const float*)d_in[12];

  // Workspace: hconv f32 (4MB) | Qbf (2MB) | Kbf (2MB) | WHbf | WLbf
  char* ws = (char*)d_ws;
  float*          hconv = (float*)ws;
  unsigned short* Qbf   = (unsigned short*)(ws + (4u << 20));
  unsigned short* Kbf   = (unsigned short*)(ws + (6u << 20));
  unsigned short* WHbf  = (unsigned short*)(ws + (8u << 20));
  unsigned short* WLbf  = (unsigned short*)(ws + (8u << 20) + (128u << 10));

  const int NW = 64 * 576;
  LLFE_wcvt<<<dim3((NW + 255) / 256), dim3(256), 0, stream>>>(wH, WHbf, NW);
  LLFE_wcvt<<<dim3((NW + 255) / 256), dim3(256), 0, stream>>>(wL, WLbf, NW);

  dim3 cgrid(64, 4), cblk(256);
  LLFE_conv_gemm<<<cgrid, cblk, 0, stream>>>(high, WHbf, gH, bH, mH, vH, hconv, Qbf);
  LLFE_conv_gemm<<<cgrid, cblk, 0, stream>>>(low,  WLbf, gL, bL, mL, vL, nullptr, Kbf);

  dim3 agrid(32, 8), ablk(256);
  LLFE_flash_attn<<<agrid, ablk, 0, stream>>>(Qbf, Kbf, hconv, p, (float*)d_out);
}
